// attention_block_64484638982870
// MI455X (gfx1250) — compile-verified, matrix operand order checked by reading, not run on hardware
//
#include <hip/hip_runtime.h>
#include <hip/hip_bf16.h>

// ---------------------------------------------------------------------------
// Attention block on MI455X (gfx1250): bf16 WMMA for every matmul,
// fp32 accumulation, flash-attention with online softmax.
// Round 2: async-LDS staging (ASYNCcnt), double-buffered GEMM pipeline,
// de-serialized B-fragment loads, hoisted V loads, global prefetch.
// ---------------------------------------------------------------------------

typedef __bf16 bf16_t;
typedef __attribute__((ext_vector_type(16))) __bf16 v16bf;
typedef __attribute__((ext_vector_type(8)))  __bf16 v8bf;
typedef __attribute__((ext_vector_type(8)))  float  v8f;
typedef __attribute__((ext_vector_type(4)))  float  v4f;

#define B_    2
#define S_    2048
#define D_    2048
#define H_    16
#define HD_   128
#define MTOT  (B_ * S_)          // 4096 rows in flattened (B*S, D)

// Flip to 0 if the async mnemonic ever fails to assemble.
#define USE_ASYNC_LDS 1

// ------------------------------ helpers ------------------------------------

__device__ __forceinline__ v16bf cat16(v8bf lo, v8bf hi) {
    return __builtin_shufflevector(lo, hi, 0, 1, 2, 3, 4, 5, 6, 7,
                                           8, 9, 10, 11, 12, 13, 14, 15);
}

__device__ __forceinline__ float redmax16(float v) {
    for (int m = 1; m < 16; m <<= 1) v = fmaxf(v, __shfl_xor(v, m, 32));
    return v;
}
__device__ __forceinline__ float redsum16(float v) {
    for (int m = 1; m < 16; m <<= 1) v += __shfl_xor(v, m, 32);
    return v;
}

// 16-byte per-lane async copy global -> LDS, tracked by ASYNCcnt.
// lds_off: wave-relative LDS byte address (low 32 bits of generic pointer,
// used as the LDS address).
__device__ __forceinline__ void async_b128(unsigned lds_off,
                                           const bf16_t* gsrc) {
#if USE_ASYNC_LDS
    asm volatile("global_load_async_to_lds_b128 %0, %1, off"
                 :
                 : "v"(lds_off), "v"(gsrc)
                 : "memory");
#else
    *(v8bf*)(uintptr_t)0 = *(const v8bf*)gsrc;  // unused in sync build
#endif
}

__device__ __forceinline__ void wait_async_zero() {
#if USE_ASYNC_LDS
    asm volatile("s_wait_asynccnt 0x0" ::: "memory");
#endif
}

// ------------------------------ cast ---------------------------------------

__global__ __launch_bounds__(256)
void cast_f32_bf16_v4(const float* __restrict__ in,
                      bf16_t* __restrict__ out, int n4) {
    int i = blockIdx.x * blockDim.x + threadIdx.x;
    if (i < n4) {
        v4f f = *(const v4f*)(in + 4 * (size_t)i);
        bf16_t* o = out + 4 * (size_t)i;
        o[0] = (bf16_t)f.x; o[1] = (bf16_t)f.y;
        o[2] = (bf16_t)f.z; o[3] = (bf16_t)f.w;
    }
}

// ------------------------------ GEMM ---------------------------------------
// C(M,N) fp32 = A(M,K) bf16 row-major  @  W(N,K)^T  (i.e. x @ w.T)
// 256 threads = 8 waves. Wave w -> 16 M-rows, full 128-col N strip.
// W tile (128 x 32 bf16 = 8 KB) double-buffered in LDS, staged with
// async global->LDS copies overlapped with the WMMA chain.

__global__ __launch_bounds__(256)
void gemm_bf16_wmma(const bf16_t* __restrict__ A,
                    const bf16_t* __restrict__ W,
                    float* __restrict__ C,
                    int M, int N, int K) {
    __shared__ bf16_t lds_w[2][128 * 32];

    const int lane = threadIdx.x & 31;
    const int wave = threadIdx.x >> 5;
    const int half = lane >> 4;     // K-half selector
    const int lm   = lane & 15;     // M (A) or N (B) index within tile

    const int n0 = blockIdx.x * 128;
    const int m0 = blockIdx.y * 128 + wave * 16;

    const int trow = threadIdx.x >> 1;        // 0..127
    const int tseg = (threadIdx.x & 1) * 16;  // 0 or 16

    v8f acc[8];
#pragma unroll
    for (int j = 0; j < 8; ++j)
#pragma unroll
        for (int e = 0; e < 8; ++e) acc[j][e] = 0.0f;

    // per-thread stage of 32B of the W tile into LDS buffer `buf`
    auto stage = [&](int buf, int k0) {
        const bf16_t* src = W + (size_t)(n0 + trow) * K + k0 + tseg;
        bf16_t* dst = &lds_w[buf][trow * 32 + tseg];
#if USE_ASYNC_LDS
        unsigned loff = (unsigned)(uintptr_t)dst;
        async_b128(loff, src);
        async_b128(loff + 16u, src + 8);
#else
        *(v8bf*)dst       = *(const v8bf*)src;
        *(v8bf*)(dst + 8) = *(const v8bf*)(src + 8);
#endif
    };

    const int nk = K >> 5;          // K/32 steps
    stage(0, 0);                    // prologue: fill buffer 0

    for (int i = 0; i < nk; ++i) {
        // tile i is in flight into buf (i&1): finish it, make visible
        wait_async_zero();
        __syncthreads();

        // kick off tile i+1 into the other buffer (overlaps compute)
        if (i + 1 < nk) stage((i + 1) & 1, (i + 1) << 5);

        // A fragment (16x32): row m = lm, K-halves per lane group
        const bf16_t* arow = A + (size_t)(m0 + lm) * K + (i << 5);
        __builtin_prefetch(arow + 32, 0, 0);     // next K-tile of A
        v16bf afrag = cat16(*(const v8bf*)(arow + half * 8),
                            *(const v8bf*)(arow + 16 + half * 8));

        // preload all 8 B fragments (one ds clause, one wait), then WMMA chain
        const bf16_t* lw = &lds_w[i & 1][0];
        v16bf bfr[8];
#pragma unroll
        for (int j = 0; j < 8; ++j) {
            const bf16_t* brow = lw + (j * 16 + lm) * 32;
            bfr[j] = cat16(*(const v8bf*)(brow + half * 8),
                           *(const v8bf*)(brow + 16 + half * 8));
        }
#pragma unroll
        for (int j = 0; j < 8; ++j)
            acc[j] = __builtin_amdgcn_wmma_f32_16x16x32_bf16(
                false, afrag, false, bfr[j], (short)0, acc[j], false, false);
    }

    // store: element (m = r + 8*half, n = lm) per C/D layout
#pragma unroll
    for (int j = 0; j < 8; ++j)
#pragma unroll
        for (int r = 0; r < 8; ++r) {
            int m = r + 8 * half;
            C[(size_t)(m0 + m) * N + n0 + j * 16 + lm] = acc[j][r];
        }
}

// ------------------------- RoPE + layout change -----------------------------
// qf/kf/vf: fp32 (B,S,D) from projections.
// qb/kb: bf16 [B,H,S,HD] (rotary applied).  vt: bf16 [B,H,HD,S] (transposed).

__global__ __launch_bounds__(256)
void rope_transpose(const float* __restrict__ qf,
                    const float* __restrict__ kf,
                    const float* __restrict__ vf,
                    const float* __restrict__ fcos,
                    const float* __restrict__ fsin,
                    bf16_t* __restrict__ qb,
                    bf16_t* __restrict__ kb,
                    bf16_t* __restrict__ vt) {
    int idx = blockIdx.x * blockDim.x + threadIdx.x;   // B*S*H*64 threads
    int j = idx & 63;                 // rotary pair index (HD/2 = 64)
    int h = (idx >> 6) & (H_ - 1);
    int s = (idx >> 10) & (S_ - 1);
    int b = idx >> 21;
    if (b >= B_) return;

    size_t src = ((size_t)b * S_ + s) * D_ + h * HD_ + 2 * j;
    float c  = fcos[s * 64 + j];
    float sn = fsin[s * 64 + j];

    float qe = qf[src], qo = qf[src + 1];
    float ke = kf[src], ko = kf[src + 1];
    float q0v = qe * c - qo * sn, q1v = qe * sn + qo * c;
    float k0v = ke * c - ko * sn, k1v = ke * sn + ko * c;

    size_t dst = (((size_t)b * H_ + h) * S_ + s) * HD_ + 2 * j;
    qb[dst]     = (bf16_t)q0v;
    qb[dst + 1] = (bf16_t)q1v;
    kb[dst]     = (bf16_t)k0v;
    kb[dst + 1] = (bf16_t)k1v;

    // V transposed: [b][h][hd][s]
    size_t vdst = (((size_t)b * H_ + h) * HD_ + 2 * j) * S_ + s;
    vt[vdst]      = (bf16_t)vf[src];
    vt[vdst + S_] = (bf16_t)vf[src + 1];
}

// ---------------------------- flash attention -------------------------------
// One wave (32 threads) per (b, h, 16-query tile). Causal, online softmax.
// Q,K: bf16 [B,H,S,HD]; Vt: bf16 [B,H,HD,S]; Oa: bf16 (B*S, D) row-major.

__global__ __launch_bounds__(32)
void flash_attn(const bf16_t* __restrict__ Q,
                const bf16_t* __restrict__ Km,
                const bf16_t* __restrict__ Vt,
                bf16_t* __restrict__ Oa) {
    __shared__ bf16_t p_lds[16 * 32];   // P tile staging (layout conversion)

    const int lane = threadIdx.x & 31;
    const int half = lane >> 4;
    const int lm   = lane & 15;

    const int q0 = blockIdx.x * 16;
    const int h  = blockIdx.y;
    const int b  = blockIdx.z;

    const size_t baseQK = ((size_t)b * H_ + h) * S_ * HD_;
    const size_t baseV  = ((size_t)b * H_ + h) * HD_ * (size_t)S_;

    // Q fragments: 4 x (16x32) covering HD = 128, resident whole loop
    v16bf qfrag[4];
    {
        const bf16_t* qrow = Q + baseQK + (size_t)(q0 + lm) * HD_;
#pragma unroll
        for (int kk = 0; kk < 4; ++kk)
            qfrag[kk] = cat16(*(const v8bf*)(qrow + kk * 32 + half * 8),
                              *(const v8bf*)(qrow + kk * 32 + 16 + half * 8));
    }

    v8f ofrag[8];
#pragma unroll
    for (int j = 0; j < 8; ++j)
#pragma unroll
        for (int e = 0; e < 8; ++e) ofrag[j][e] = 0.0f;

    float rmax[8], rsum[8];
#pragma unroll
    for (int r = 0; r < 8; ++r) { rmax[r] = -1e30f; rsum[r] = 0.0f; }

    const float scale = 0.088388347648318447f;   // 1/sqrt(128)
    const int kend = q0 + 16;                    // causal bound (exclusive)

    for (int kb = 0; kb < kend; kb += 32) {
        // ---- scores: two 16x16 tiles (keys kb..kb+15, kb+16..kb+31)
        v8f sc[2];
#pragma unroll
        for (int t = 0; t < 2; ++t) {
#pragma unroll
            for (int e = 0; e < 8; ++e) sc[t][e] = 0.0f;
            int krow = kb + t * 16 + lm;
            if (krow > S_ - 1) krow = S_ - 1;            // clamp tail reads
            const bf16_t* kr = Km + baseQK + (size_t)krow * HD_;
#pragma unroll
            for (int kk = 0; kk < 4; ++kk) {
                v16bf bfrag = cat16(*(const v8bf*)(kr + kk * 32 + half * 8),
                                    *(const v8bf*)(kr + kk * 32 + 16 + half * 8));
                sc[t] = __builtin_amdgcn_wmma_f32_16x16x32_bf16(
                    false, qfrag[kk], false, bfrag, (short)0, sc[t], false, false);
            }
        }

        // ---- hoist V fragment loads: latency hidden under softmax math
        v16bf vfr[8];
#pragma unroll
        for (int j = 0; j < 8; ++j) {
            const bf16_t* vr = Vt + baseV + (size_t)(j * 16 + lm) * S_ + kb;
            vfr[j] = cat16(*(const v8bf*)(vr + half * 8),
                           *(const v8bf*)(vr + 16 + half * 8));
        }

        // ---- scale + causal mask + online softmax
        float p0[8], p1[8], alpha[8];
#pragma unroll
        for (int r = 0; r < 8; ++r) {
            int qrow = q0 + r + 8 * half;
            int key0 = kb + lm;
            int key1 = kb + 16 + lm;
            float s0 = sc[0][r] * scale + (key0 > qrow ? -1e9f : 0.0f);
            float s1 = sc[1][r] * scale + (key1 > qrow ? -1e9f : 0.0f);
            float mx = redmax16(fmaxf(s0, s1));
            float mnew = fmaxf(rmax[r], mx);
            alpha[r] = __expf(rmax[r] - mnew);
            rmax[r] = mnew;
            float e0 = __expf(s0 - mnew);
            float e1 = __expf(s1 - mnew);
            rsum[r] = rsum[r] * alpha[r] + redsum16(e0 + e1);
            p0[r] = e0;
            p1[r] = e1;
        }

        // ---- rescale running output
#pragma unroll
        for (int j = 0; j < 8; ++j)
#pragma unroll
            for (int r = 0; r < 8; ++r) ofrag[j][r] *= alpha[r];

        // ---- P: C/D layout -> A-fragment layout via LDS round trip
#pragma unroll
        for (int r = 0; r < 8; ++r) {
            int m = r + 8 * half;
            p_lds[m * 32 + lm]      = (bf16_t)p0[r];
            p_lds[m * 32 + 16 + lm] = (bf16_t)p1[r];
        }
        __syncthreads();
        v16bf pfrag = cat16(*(const v8bf*)&p_lds[lm * 32 + half * 8],
                            *(const v8bf*)&p_lds[lm * 32 + 16 + half * 8]);
        __syncthreads();

        // ---- P(16x32) @ V(32x128): 8 WMMA over HD tiles
#pragma unroll
        for (int j = 0; j < 8; ++j)
            ofrag[j] = __builtin_amdgcn_wmma_f32_16x16x32_bf16(
                false, pfrag, false, vfr[j], (short)0, ofrag[j], false, false);
    }

    // ---- normalize and store attn output as bf16 (B*S, D) row-major
#pragma unroll
    for (int j = 0; j < 8; ++j)
#pragma unroll
        for (int r = 0; r < 8; ++r) {
            int m = r + 8 * half;
            float val = ofrag[j][r] / rsum[r];
            Oa[((size_t)b * S_ + q0 + m) * D_ + h * HD_ + j * 16 + lm] =
                (bf16_t)val;
        }
}

// ------------------------------ launcher ------------------------------------

extern "C" void kernel_launch(void* const* d_in, const int* in_sizes, int n_in,
                              void* d_out, int out_size, void* d_ws, size_t ws_size,
                              hipStream_t stream) {
    const float* x    = (const float*)d_in[0];
    const float* fcos = (const float*)d_in[2];
    const float* fsin = (const float*)d_in[3];
    const float* wq   = (const float*)d_in[5];
    const float* wk   = (const float*)d_in[6];
    const float* wv   = (const float*)d_in[7];
    const float* wo   = (const float*)d_in[8];
    float* out = (float*)d_out;

    const size_t MD = (size_t)MTOT * D_;   // 8,388,608 elems
    const size_t DD = (size_t)D_ * D_;     // 4,194,304 elems

    char* ws = (char*)d_ws;
    size_t off = 0;
    auto carve = [&](size_t bytes) -> char* {
        char* p = ws + off;
        off += (bytes + 255) & ~(size_t)255;
        return p;
    };

    bf16_t* xb  = (bf16_t*)carve(MD * 2);
    bf16_t* wqb = (bf16_t*)carve(DD * 2);
    bf16_t* wkb = (bf16_t*)carve(DD * 2);
    bf16_t* wvb = (bf16_t*)carve(DD * 2);
    bf16_t* wob = (bf16_t*)carve(DD * 2);
    float*  qf  = (float*)carve(MD * 4);
    float*  kf  = (float*)carve(MD * 4);
    float*  vf  = (float*)carve(MD * 4);
    bf16_t* qb  = (bf16_t*)carve(MD * 2);
    bf16_t* kb  = (bf16_t*)carve(MD * 2);
    bf16_t* vt  = (bf16_t*)carve(MD * 2 + 4096);  // pad: tail key-block slack
    bf16_t* atb = (bf16_t*)carve(MD * 2);

    // 1) downcast inputs (vectorized x4)
    cast_f32_bf16_v4<<<(int)(MD / 4 / 256), 256, 0, stream>>>(x, xb, (int)(MD / 4));
    cast_f32_bf16_v4<<<(int)(DD / 4 / 256), 256, 0, stream>>>(wq, wqb, (int)(DD / 4));
    cast_f32_bf16_v4<<<(int)(DD / 4 / 256), 256, 0, stream>>>(wk, wkb, (int)(DD / 4));
    cast_f32_bf16_v4<<<(int)(DD / 4 / 256), 256, 0, stream>>>(wv, wvb, (int)(DD / 4));
    cast_f32_bf16_v4<<<(int)(DD / 4 / 256), 256, 0, stream>>>(wo, wob, (int)(DD / 4));

    // 2) QKV projections: (4096,2048) = xb @ W^T
    dim3 ggrid(D_ / 128, MTOT / 128, 1);
    gemm_bf16_wmma<<<ggrid, 256, 0, stream>>>(xb, wqb, qf, MTOT, D_, D_);
    gemm_bf16_wmma<<<ggrid, 256, 0, stream>>>(xb, wkb, kf, MTOT, D_, D_);
    gemm_bf16_wmma<<<ggrid, 256, 0, stream>>>(xb, wvb, vf, MTOT, D_, D_);

    // 3) RoPE + relayout (Q,K -> [B,H,S,HD] bf16; V -> [B,H,HD,S] bf16)
    const int rope_threads = B_ * S_ * H_ * (HD_ / 2);   // 4,194,304
    rope_transpose<<<rope_threads / 256, 256, 0, stream>>>(
        qf, kf, vf, fcos, fsin, qb, kb, vt);

    // 4) causal flash attention, wave per (b,h,16 queries)
    dim3 agrid(S_ / 16, H_, B_);
    flash_attn<<<agrid, 32, 0, stream>>>(qb, kb, vt, atb);

    // 5) output projection straight into d_out (fp32)
    gemm_bf16_wmma<<<ggrid, 256, 0, stream>>>(atb, wob, out, MTOT, D_, D_);
}
